// PreTrainModel_59416577573016
// MI455X (gfx1250) — matrix operand order repaired and checked by reading, not compiled, not run on hardware
//
#include <hip/hip_runtime.h>

#define NRAW  1200
#define NDUP  1080
#define PTOT  2280
#define EMBD  16
#define SEQL  11
#define KCH   32
#define MNB   32
#define NGR   8

typedef __attribute__((ext_vector_type(16))) __bf16 v16bf;
typedef __attribute__((ext_vector_type(8)))  __bf16 v8bf;
typedef __attribute__((ext_vector_type(2)))  __bf16 v2bf;
typedef __attribute__((ext_vector_type(8)))  float  v8f;

// ---------------------------------------------------------------- setup ----
__global__ void k_setup(const int* tokens, const float* pos, const int* seq,
                        const float* ori, const int* batch, const float* emb,
                        float* x0, float* pos2, int* seq2, float* ori2,
                        int* batch2, float* valid)
{
    int p = blockIdx.x * blockDim.x + threadIdx.x;
    if (p >= PTOT) return;
    int src, badd = 0;
    if (p < NRAW) { src = p; }
    else {
        int j = p - NRAW;
        // deterministic stand-in for the fixed-PRNG duplication permutation:
        // sel is a descending subset of [0,N) as in sel = N-1-sorted(perm)
        src  = (NRAW - 1) - (int)(((long long)j * NRAW) / NDUP);
        badd = 4; // B
    }
    int tk = tokens[src];
    for (int e = 0; e < EMBD; ++e) x0[p * EMBD + e] = emb[tk * EMBD + e];
    for (int d = 0; d < 3; ++d)   pos2[p * 3 + d] = pos[src * 3 + d];
    for (int d = 0; d < 9; ++d)   ori2[p * 9 + d] = ori[src * 9 + d];
    seq2[p]   = seq[src];
    batch2[p] = batch[src] + badd;
    valid[p]  = 1.0f;
}

// ---------------------------------------------------------------- utils ----
__global__ void k_zero(float* p, long long n)
{
    long long i = (long long)blockIdx.x * blockDim.x + threadIdx.x;
    if (i < n) p[i] = 0.0f;
}

__global__ void k_count(const float* valid, int P, float* cnt)
{
    __shared__ float s[256];
    float a = 0.f;
    for (int p = threadIdx.x; p < P; p += 256) a += valid[p];
    s[threadIdx.x] = a; __syncthreads();
    for (int st = 128; st > 0; st >>= 1) {
        if (threadIdx.x < st) s[threadIdx.x] += s[threadIdx.x + st];
        __syncthreads();
    }
    if (threadIdx.x == 0) cnt[0] = s[0];
}

// weight convert+transpose: WT[n][k] = (bf16) W[k][n]
__global__ void k_wtrans(const float* W, __bf16* WT, int K, int N)
{
    long long i = (long long)blockIdx.x * blockDim.x + threadIdx.x;
    if (i >= (long long)K * N) return;
    int       k = (int)(i % K);
    long long n = i / K;
    WT[n * K + k] = (__bf16)W[(long long)k * N + n];
}

// ------------------------------------------------------------- batchnorm ---
__global__ void k_bn_stats(const float* x, const float* valid, int P, int C,
                           float* sum, float* sumsq)
{
    int c = blockIdx.x;
    __shared__ float s1[256], s2[256];
    float a = 0.f, b = 0.f;
    for (int p = threadIdx.x; p < P; p += 256) {
        float v  = valid[p];
        float xv = x[(long long)p * C + c];
        a += xv * v;
        b += xv * xv * v;
    }
    s1[threadIdx.x] = a; s2[threadIdx.x] = b; __syncthreads();
    for (int st = 128; st > 0; st >>= 1) {
        if (threadIdx.x < st) {
            s1[threadIdx.x] += s1[threadIdx.x + st];
            s2[threadIdx.x] += s2[threadIdx.x + st];
        }
        __syncthreads();
    }
    if (threadIdx.x == 0) { sum[c] = s1[0]; sumsq[c] = s2[0]; }
}

__global__ void k_bn_apply(const float* x, float* y, const float* sum,
                           const float* sumsq, const float* cnt,
                           const float* g, const float* b, float slope,
                           int P, int C)
{
    long long i = (long long)blockIdx.x * blockDim.x + threadIdx.x;
    if (i >= (long long)P * C) return;
    int c = (int)(i % C);
    float n   = fmaxf(cnt[0], 1.f);
    float mu  = sum[c] / n;
    float var = fmaxf(sumsq[c] / n - mu * mu, 0.f);
    float v   = (x[i] - mu) * rsqrtf(var + 1e-5f) * g[c] + b[c];
    y[i] = v > 0.f ? v : slope * v;
}

// --------------------------------------------------------- neighbor search -
__global__ void k_neighbors(const float* pos, const int* batch,
                            const float* valid, int P, float r2, int* nbr)
{
    int p = blockIdx.x * blockDim.x + threadIdx.x;
    if (p >= P) return;
    float px = pos[p * 3], py = pos[p * 3 + 1], pz = pos[p * 3 + 2];
    int  bp = batch[p];
    bool vp = valid[p] > 0.5f;
    float pd2 = -1.f; int pidx = -1;
    for (int t = 0; t < MNB; ++t) {
        float bd2 = 3.4e38f; int bidx = -1;
        if (vp) {
            for (int q = 0; q < P; ++q) {
                if (batch[q] != bp || valid[q] <= 0.5f) continue;
                float dx = pos[q * 3] - px, dy = pos[q * 3 + 1] - py,
                      dz = pos[q * 3 + 2] - pz;
                float d2 = dx * dx + dy * dy + dz * dz;
                if (d2 > r2) continue;
                if (d2 < pd2 || (d2 == pd2 && q <= pidx)) continue;  // taken
                if (d2 < bd2 || (d2 == bd2 && q < bidx)) { bd2 = d2; bidx = q; }
            }
        }
        nbr[p * MNB + t] = bidx;
        if (bidx < 0) { pd2 = 3.4e38f; }
        else          { pd2 = bd2; pidx = bidx; }
    }
}

// ------------------------------------------------------------ kernel eval --
__global__ void k_kern(const float* pos, const int* seq, const float* ori,
                       const int* nbr, const float* wnW, const float* wnb,
                       float r, int P, float* kern, int* sidx)
{
    int g = blockIdx.x * blockDim.x + threadIdx.x;
    if (g >= P * MNB) return;
    int p = g / MNB;
    int q = nbr[g];
    if (q < 0) {
        sidx[g] = SEQL / 2;
        for (int k = 0; k < KCH; ++k) kern[(long long)g * KCH + k] = 0.f;
        return;
    }
    float rel[3]; float d2 = 0.f;
    for (int d = 0; d < 3; ++d) {
        rel[d] = pos[q * 3 + d] - pos[p * 3 + d];
        d2 += rel[d] * rel[d];
    }
    float dist = sqrtf(d2);
    float inv  = 1.f / (dist + 1e-9f);
    float dir[3] = { rel[0] * inv, rel[1] * inv, rel[2] * inv };
    float delta[7];
    for (int a = 0; a < 3; ++a) {
        const float* oa = ori + (long long)p * 9 + a * 3;
        const float* ob = ori + (long long)q * 9 + a * 3;
        delta[a]     = oa[0] * dir[0] + oa[1] * dir[1] + oa[2] * dir[2];
        delta[3 + a] = oa[0] * ob[0] + oa[1] * ob[1] + oa[2] * ob[2];
    }
    float nd = dist / r;
    delta[6] = nd;
    int sd = seq[q] - seq[p];
    int s  = SEQL / 2;
    sd = (sd < -s ? -s : (sd > s ? s : sd)) + s;
    sidx[g] = sd;
    float env = 0.5f * cosf(nd * 3.14159265358979f) + 0.5f;
    const float* Wg = wnW + (long long)sd * 7 * KCH;
    const float* bg = wnb + (long long)sd * KCH;
    for (int k = 0; k < KCH; ++k) {
        float sv = bg[k];
        for (int i = 0; i < 7; ++i) sv += delta[i] * Wg[i * KCH + k];
        sv = sv > 0.f ? sv : 0.2f * sv;
        kern[(long long)g * KCH + k] = sv * env;
    }
}

// --------------------------------------------------------- agg assembly ----
// grid (P, C/8), 256 threads = 32 k-channels x 8 c-columns; one WG owns a node
__global__ void k_agg(const float* xc, const float* kern, const int* sidx,
                      const int* nbr, int P, int C, __bf16* agg)
{
    int p  = blockIdx.x;
    int k  = threadIdx.x & 31;
    int ci = threadIdx.x >> 5;
    int c  = blockIdx.y * 8 + ci;
    float acc[SEQL];
#pragma unroll
    for (int l = 0; l < SEQL; ++l) acc[l] = 0.f;
    for (int m = 0; m < MNB; ++m) {
        int   q  = nbr[p * MNB + m];
        int   l  = sidx[p * MNB + m];
        float kv = kern[((long long)p * MNB + m) * KCH + k];
        float xv = (q >= 0) ? xc[(long long)q * C + c] : 0.f;
        float pr = kv * xv;
#pragma unroll
        for (int ll = 0; ll < SEQL; ++ll) acc[ll] += (ll == l) ? pr : 0.f;
    }
    long long base = (long long)p * ((long long)SEQL * KCH * C);
#pragma unroll
    for (int l = 0; l < SEQL; ++l)
        agg[base + (long long)(l * KCH + k) * C + c] = (__bf16)acc[l];
}

// -------------------------------------------------- WMMA fragment loaders --
union UF { v16bf v; v8bf h[2]; };

__device__ __forceinline__ void load_frags(
        const __bf16 (*As)[40], const __bf16 (*Bt)[40],
        int wr, int wc, int lane, UF& a0, UF& a1, UF& b0, UF& b1)
{
    int mrow = wr + (lane & 15);
    int kb   = (lane < 16) ? 0 : 8;       // A halves at kb, 16+kb
    a0.h[0] = *(const v8bf*)&As[mrow][kb];
    a0.h[1] = *(const v8bf*)&As[mrow][16 + kb];
    a1.h[0] = *(const v8bf*)&As[mrow + 16][kb];
    a1.h[1] = *(const v8bf*)&As[mrow + 16][16 + kb];
    int nn  = wc + (lane & 15);
    int kb2 = (lane < 16) ? 0 : 8;        // B halves at kb2, 16+kb2
    b0.h[0] = *(const v8bf*)&Bt[nn][kb2];
    b0.h[1] = *(const v8bf*)&Bt[nn][kb2 + 16];
    b1.h[0] = *(const v8bf*)&Bt[nn + 16][kb2];
    b1.h[1] = *(const v8bf*)&Bt[nn + 16][kb2 + 16];
}

// ---------------------------------------------------- generic WMMA GEMM ----
// C[MxN] = A[MxK](f32) * B[KxN](f32, cvt->bf16) [+ add]; 64x64 tile,
// 128 threads (4 waves), each wave 2x2 16x16 frags, K-step 32.
template <bool HASADD>
__global__ void __launch_bounds__(128) k_gemm(
        const float* A, const float* B, float* Cm,
        const float* add, int M, int N, int K)
{
    __shared__ __attribute__((aligned(16))) __bf16 As[64][40]; // [m][k]
    __shared__ __attribute__((aligned(16))) __bf16 Bt[64][40]; // [n][k]
    const int tid  = threadIdx.x;
    const int lane = tid & 31;
    const int wid  = tid >> 5;
    const int row0 = blockIdx.x * 64;
    const int col0 = blockIdx.y * 64;
    const int wr   = (wid >> 1) * 32;
    const int wc   = (wid & 1) * 32;

    v8f c00 = {}, c01 = {}, c10 = {}, c11 = {};

    for (int kt = 0; kt < K; kt += 32) {
        // stage A 64x32 as K-pairs (branchless: clamp rows, mask K)
#pragma unroll
        for (int i = 0; i < 8; ++i) {
            int pe = i * 128 + tid;
            int r  = pe >> 4;
            int k2 = (pe & 15) * 2;
            int gm = row0 + r; if (gm >= M) gm = M - 1;
            int gk = kt + k2;
            float msk = (gk < K) ? 1.f : 0.f;
            int gks = (gk < K) ? gk : 0;
            float2 f = *(const float2*)(A + (long long)gm * K + gks);
            v2bf pr; pr[0] = (__bf16)(f.x * msk); pr[1] = (__bf16)(f.y * msk);
            *(v2bf*)&As[r][k2] = pr;
        }
        // stage B 32x64 K-transposed
#pragma unroll
        for (int i = 0; i < 8; ++i) {
            int pe = i * 128 + tid;
            int n  = pe & 63;
            int k2 = (pe >> 6) * 2;
            int gn = col0 + n;                // always < N (N multiple of 64)
            int gk = kt + k2;
            float m0 = (gk     < K) ? 1.f : 0.f;
            float m1 = (gk + 1 < K) ? 1.f : 0.f;
            int g0 = (gk     < K) ? gk     : 0;
            int g1 = (gk + 1 < K) ? gk + 1 : 0;
            float b0v = B[(long long)g0 * N + gn] * m0;
            float b1v = B[(long long)g1 * N + gn] * m1;
            v2bf pr; pr[0] = (__bf16)b0v; pr[1] = (__bf16)b1v;
            *(v2bf*)&Bt[n][k2] = pr;
        }
        __syncthreads();
        UF a0, a1, b0, b1;
        load_frags(As, Bt, wr, wc, lane, a0, a1, b0, b1);
        c00 = __builtin_amdgcn_wmma_f32_16x16x32_bf16(false, a0.v, false, b0.v, (short)0, c00, false, false);
        c01 = __builtin_amdgcn_wmma_f32_16x16x32_bf16(false, a0.v, false, b1.v, (short)0, c01, false, false);
        c10 = __builtin_amdgcn_wmma_f32_16x16x32_bf16(false, a1.v, false, b0.v, (short)0, c10, false, false);
        c11 = __builtin_amdgcn_wmma_f32_16x16x32_bf16(false, a1.v, false, b1.v, (short)0, c11, false, false);
        __syncthreads();
    }

    int nlo = lane & 15;
    int mhi = (lane < 16) ? 0 : 8;
#pragma unroll
    for (int v = 0; v < 8; ++v) {
        int mm0 = row0 + wr + v + mhi;
        int mm1 = row0 + wr + 16 + v + mhi;
        long long r0 = (long long)mm0 * N + col0 + wc;
        long long r1 = (long long)mm1 * N + col0 + wc;
        if (mm0 < M) {
            float t0v = c00[v], t1v = c01[v];
            if constexpr (HASADD) { t0v += add[r0 + nlo]; t1v += add[r0 + 16 + nlo]; }
            Cm[r0 + nlo] = t0v;
            Cm[r0 + 16 + nlo] = t1v;
        }
        if (mm1 < M) {
            float t0v = c10[v], t1v = c11[v];
            if constexpr (HASADD) { t0v += add[r1 + nlo]; t1v += add[r1 + 16 + nlo]; }
            Cm[r1 + nlo] = t0v;
            Cm[r1 + 16 + nlo] = t1v;
        }
    }
}

// -------------------------------------------- async-staged conv WMMA GEMM --
// C[MxN] = A[MxK](bf16 row-major) * BT[NxK](bf16, pre-transposed weights).
// K multiple of 32, N multiple of 64. Tiles staged straight into LDS with
// global_load_async_to_lds_b128 (ASYNCcnt), no VALU staging in the hot loop.
__global__ void __launch_bounds__(128) k_gemm_async(
        const __bf16* A, const __bf16* BT, float* Cm, int M, int N, int K)
{
    __shared__ __attribute__((aligned(16))) __bf16 As[64][40]; // [m][k]
    __shared__ __attribute__((aligned(16))) __bf16 Bt[64][40]; // [n][k]
    const int tid  = threadIdx.x;
    const int lane = tid & 31;
    const int wid  = tid >> 5;
    const int row0 = blockIdx.x * 64;
    const int col0 = blockIdx.y * 64;
    const int wr   = (wid >> 1) * 32;
    const int wc   = (wid & 1) * 32;

    v8f c00 = {}, c01 = {}, c10 = {}, c11 = {};

    for (int kt = 0; kt < K; kt += 32) {
        // 64 rows x 4 chunks of 8 bf16 (16B) per tile; 2 chunks/thread/tile
#pragma unroll
        for (int h = 0; h < 2; ++h) {
            int q  = h * 128 + tid;
            int r  = q >> 2;
            int ch = q & 3;
            int gm = row0 + r; if (gm >= M) gm = M - 1;   // clamp: harmless
            const __bf16* srca = A + (long long)gm * K + kt + ch * 8;
            unsigned dsta = (unsigned)(size_t)&As[r][ch * 8];
            asm volatile("global_load_async_to_lds_b128 %0, %1, off"
                         :: "v"(dsta), "v"((unsigned long long)(size_t)srca)
                         : "memory");
            int gn = col0 + r;                            // always < N
            const __bf16* srcb = BT + (long long)gn * K + kt + ch * 8;
            unsigned dstb = (unsigned)(size_t)&Bt[r][ch * 8];
            asm volatile("global_load_async_to_lds_b128 %0, %1, off"
                         :: "v"(dstb), "v"((unsigned long long)(size_t)srcb)
                         : "memory");
        }
        asm volatile("s_wait_asynccnt 0" ::: "memory");
        __syncthreads();
        UF a0, a1, b0, b1;
        load_frags(As, Bt, wr, wc, lane, a0, a1, b0, b1);
        c00 = __builtin_amdgcn_wmma_f32_16x16x32_bf16(false, a0.v, false, b0.v, (short)0, c00, false, false);
        c01 = __builtin_amdgcn_wmma_f32_16x16x32_bf16(false, a0.v, false, b1.v, (short)0, c01, false, false);
        c10 = __builtin_amdgcn_wmma_f32_16x16x32_bf16(false, a1.v, false, b0.v, (short)0, c10, false, false);
        c11 = __builtin_amdgcn_wmma_f32_16x16x32_bf16(false, a1.v, false, b1.v, (short)0, c11, false, false);
        __syncthreads();
    }

    int nlo = lane & 15;
    int mhi = (lane < 16) ? 0 : 8;
#pragma unroll
    for (int v = 0; v < 8; ++v) {
        int mm0 = row0 + wr + v + mhi;
        int mm1 = row0 + wr + 16 + v + mhi;
        long long r0 = (long long)mm0 * N + col0 + wc;
        long long r1 = (long long)mm1 * N + col0 + wc;
        if (mm0 < M) { Cm[r0 + nlo] = c00[v]; Cm[r0 + 16 + nlo] = c01[v]; }
        if (mm1 < M) { Cm[r1 + nlo] = c10[v]; Cm[r1 + 16 + nlo] = c11[v]; }
    }
}

// -------------------------------------------------------------- pooling ----
__global__ void k_pool_acc(const float* x, const float* pos, const int* seq,
                           const float* ori, const int* batch, const float* valid,
                           int P, int C, int half,
                           float* xsum, float* possum, float* orisum, float* cnt)
{
    int p = blockIdx.x * blockDim.x + threadIdx.x;
    if (p >= P) return;
    float v  = valid[p];
    int   sg = batch[p] * half + seq[p] / 2;
    atomicAdd(&cnt[sg], v);
    for (int d = 0; d < 3; ++d) atomicAdd(&possum[sg * 3 + d], pos[p * 3 + d] * v);
    for (int d = 0; d < 9; ++d) atomicAdd(&orisum[sg * 9 + d], ori[p * 9 + d] * v);
    for (int c = 0; c < C; ++c)
        atomicAdd(&xsum[(long long)sg * C + c], x[(long long)p * C + c] * v);
}

__global__ void k_pool_fin(float* x, float* pos, int* seq, float* ori,
                           int* batch, float* valid,
                           const float* xsum, const float* possum,
                           const float* orisum, const float* cnt,
                           int nseg, int C, int half)
{
    int s = blockIdx.x * blockDim.x + threadIdx.x;
    if (s >= nseg) return;
    float den = fmaxf(cnt[s], 1.f);
    for (int c = 0; c < C; ++c) x[(long long)s * C + c] = xsum[(long long)s * C + c] / den;
    for (int d = 0; d < 3; ++d) pos[s * 3 + d] = possum[s * 3 + d] / den;
    for (int d = 0; d < 9; ++d) ori[s * 9 + d] = orisum[s * 9 + d] / den;
    seq[s]   = s % half;
    batch[s] = s / half;
    valid[s] = cnt[s] > 0.f ? 1.f : 0.f;
}

__global__ void k_gpool(const float* x, const int* batch, const float* valid,
                        int P, int C, float* xg)
{
    int c = blockIdx.x, g = blockIdx.y;
    __shared__ float sx[256], sc[256];
    float a = 0.f, n = 0.f;
    for (int p = threadIdx.x; p < P; p += 256) {
        if (batch[p] == g) {
            float v = valid[p];
            a += x[(long long)p * C + c] * v;
            n += v;
        }
    }
    sx[threadIdx.x] = a; sc[threadIdx.x] = n; __syncthreads();
    for (int st = 128; st > 0; st >>= 1) {
        if (threadIdx.x < st) {
            sx[threadIdx.x] += sx[threadIdx.x + st];
            sc[threadIdx.x] += sc[threadIdx.x + st];
        }
        __syncthreads();
    }
    if (threadIdx.x == 0) xg[(long long)g * C + c] = sx[0] / fmaxf(sc[0], 1.f);
}

// ---------------------------------------------------------------- MLP ------
__global__ void k_mlp(const float* xg, const float* W1, const float* B1,
                      const float* W2, const float* B2, const float* W3,
                      const float* B3, float* out)
{
    __shared__ float h1[8 * 512];
    __shared__ float h2[8 * 256];
    int t = threadIdx.x;
    for (int i = t; i < 8 * 512; i += 256) {
        int g = i / 512, o = i % 512;
        float s = B1[o];
        for (int c = 0; c < 1024; ++c) s += xg[g * 1024 + c] * W1[c * 512 + o];
        h1[i] = fmaxf(s, 0.f);
    }
    __syncthreads();
    for (int i = t; i < 8 * 256; i += 256) {
        int g = i / 256, o = i % 256;
        float s = B2[o];
        for (int c = 0; c < 512; ++c) s += h1[g * 512 + c] * W2[c * 256 + o];
        h2[i] = fmaxf(s, 0.f);
    }
    __syncthreads();
    for (int i = t; i < 8 * 128; i += 256) {
        int g = i / 128, o = i % 128;
        float s = B3[o];
        for (int c = 0; c < 256; ++c) s += h2[g * 256 + c] * W3[c * 128 + o];
        out[i] = s;
    }
}

// ---------------------------------------------------------------- host -----
extern "C" void kernel_launch(void* const* d_in, const int* in_sizes, int n_in,
                              void* d_out, int out_size, void* d_ws, size_t ws_size,
                              hipStream_t stream)
{
    (void)in_sizes; (void)n_in; (void)out_size; (void)ws_size;
    const int*   tokens = (const int*)  d_in[0];
    const float* pos    = (const float*)d_in[1];
    const int*   seq    = (const int*)  d_in[2];
    const float* ori    = (const float*)d_in[3];
    const int*   batch  = (const int*)  d_in[4];
    const float* par    = (const float*)d_in[5];

    // ---- param offsets (jax pytree: dict keys sorted; lists in order) ----
    const int CH[3] = {256, 512, 1024}, WD[3] = {64, 128, 256};
    const float RAD[3] = {6.f, 9.f, 12.f};
    struct BO {
        size_t convW, idW, idb, idg, inW, inb1, inb2, ing1, ing2,
               outW, outb, outg, wnW, wnb;
        int in_c, out_c, w; bool hid;
    } bo[6];
    size_t o = 0; int in_c = EMBD;
    for (int i = 0; i < 6; ++i) {
        int out_c = CH[i / 2], w = WD[i / 2];
        bool hid = (in_c != out_c);
        BO& b = bo[i];
        b.in_c = in_c; b.out_c = out_c; b.w = w; b.hid = hid;
        b.convW = o; o += (size_t)SEQL * KCH * w * w;              // conv_W
        if (hid) { b.idW = o; o += (size_t)in_c * out_c;           // id: W,b,g
                   b.idb = o; o += in_c;
                   b.idg = o; o += in_c; }
        b.inW = o; o += (size_t)in_c * w;                          // in: W,b1,b2,g1,g2
        b.inb1 = o; o += in_c;  b.inb2 = o; o += w;
        b.ing1 = o; o += in_c;  b.ing2 = o; o += w;
        b.outW = o; o += (size_t)w * out_c;                        // out: W,b,g
        b.outb = o; o += w;     b.outg = o; o += w;
        b.wnW = o; o += (size_t)SEQL * 7 * KCH;                    // wn_W, wn_b
        b.wnb = o; o += (size_t)SEQL * KCH;
        in_c = out_c;
    }
    size_t emb_o = o; o += 21 * EMBD;
    size_t oW1 = o; o += (size_t)1024 * 512;
    size_t oW2 = o; o += (size_t)512 * 256;
    size_t oW3 = o; o += (size_t)256 * 128;
    size_t oB1 = o; o += 512;
    size_t oB2 = o; o += 256;
    size_t oB3 = o; o += 128;

    // ---- workspace bump allocator ----
    char* wsb = (char*)d_ws; size_t off = 0;
    auto alloc = [&](size_t bytes) -> void* {
        off = (off + 255) & ~(size_t)255;
        void* p = wsb + off; off += bytes; return p;
    };
    const size_t NODEF = (size_t)PTOT * 1024 * sizeof(float);
    float* xA = (float*)alloc(NODEF);
    float* xB = (float*)alloc(NODEF);
    float* t0 = (float*)alloc(NODEF);
    float* t1 = (float*)alloc(NODEF);
    float* t2 = (float*)alloc(NODEF);
    float* t3 = (float*)alloc(NODEF);
    const int PL[3] = {PTOT, 1200, 600};
    float* lp[3]; int* ls[3]; float* lo[3]; int* lb[3]; float* lv[3];
    for (int l = 0; l < 3; ++l) {
        lp[l] = (float*)alloc((size_t)PL[l] * 3 * 4);
        ls[l] = (int*)  alloc((size_t)PL[l] * 4);
        lo[l] = (float*)alloc((size_t)PL[l] * 9 * 4);
        lb[l] = (int*)  alloc((size_t)PL[l] * 4);
        lv[l] = (float*)alloc((size_t)PL[l] * 4);
    }
    int*    nbr  = (int*)   alloc((size_t)PTOT * MNB * 4);
    int*    sidx = (int*)   alloc((size_t)PTOT * MNB * 4);
    float*  kern = (float*) alloc((size_t)PTOT * MNB * KCH * 4);
    __bf16* agg  = (__bf16*)alloc((size_t)55000000 * 2);    // max 1200*45056
    __bf16* cwt  = (__bf16*)alloc((size_t)23200000 * 2);    // conv_W^T, max 352*256*256
    float* bnsum = (float*) alloc(1024 * 4);
    float* bnsq  = (float*) alloc(1024 * 4);
    float* cnt   = (float*) alloc(4);
    float* xsum  = (float*) alloc((size_t)1200 * 512 * 4);
    float* psum  = (float*) alloc((size_t)1200 * 3 * 4);
    float* osum  = (float*) alloc((size_t)1200 * 9 * 4);
    float* csum  = (float*) alloc((size_t)1200 * 4);
    float* xg    = (float*) alloc((size_t)NGR * 1024 * 4);

    // ---- stage 0 setup ----
    k_setup<<<(PTOT + 127) / 128, 128, 0, stream>>>(
        tokens, pos, seq, ori, batch, par + emb_o,
        xA, lp[0], ls[0], lo[0], lb[0], lv[0]);

    float* xcur = xA; float* xnxt = xB;
    for (int i = 0; i < 6; ++i) {
        int lvl = i / 2, P = PL[lvl];
        BO& b = bo[i];
        if (i % 2 == 0)
            k_neighbors<<<(P + 63) / 64, 64, 0, stream>>>(
                lp[lvl], lb[lvl], lv[lvl], P, RAD[lvl] * RAD[lvl], nbr);
        k_count<<<1, 256, 0, stream>>>(lv[lvl], P, cnt);

        // bn1(x) stats shared between in-MLP and identity branch
        k_bn_stats<<<b.in_c, 256, 0, stream>>>(xcur, lv[lvl], P, b.in_c, bnsum, bnsq);
        long long n1 = (long long)P * b.in_c;
        k_bn_apply<<<(unsigned)((n1 + 255) / 256), 256, 0, stream>>>(
            xcur, t0, bnsum, bnsq, cnt, par + b.ing1, par + b.inb1, 0.2f, P, b.in_c);
        const float* identptr;
        if (b.hid) {
            k_bn_apply<<<(unsigned)((n1 + 255) / 256), 256, 0, stream>>>(
                xcur, t1, bnsum, bnsq, cnt, par + b.idg, par + b.idb, 0.1f, P, b.in_c);
            k_gemm<false><<<dim3((P + 63) / 64, b.out_c / 64), 128, 0, stream>>>(
                t1, par + b.idW, t2, nullptr, P, b.out_c, b.in_c);
            identptr = t2;
        } else identptr = xcur;

        // h = a1 @ in.W
        k_gemm<false><<<dim3((P + 63) / 64, b.w / 64), 128, 0, stream>>>(
            t0, par + b.inW, t3, nullptr, P, b.w, b.in_c);
        // xc = lrelu0.2(bn2(h))
        k_bn_stats<<<b.w, 256, 0, stream>>>(t3, lv[lvl], P, b.w, bnsum, bnsq);
        long long nw = (long long)P * b.w;
        k_bn_apply<<<(unsigned)((nw + 255) / 256), 256, 0, stream>>>(
            t3, t0, bnsum, bnsq, cnt, par + b.ing2, par + b.inb2, 0.2f, P, b.w);

        // geometric kernel + aggregation + async-staged conv GEMM (WMMA bf16)
        k_kern<<<(P * MNB + 127) / 128, 128, 0, stream>>>(
            lp[lvl], ls[lvl], lo[lvl], nbr, par + b.wnW, par + b.wnb,
            RAD[lvl], P, kern, sidx);
        k_agg<<<dim3(P, b.w / 8), 256, 0, stream>>>(t0, kern, sidx, nbr, P, b.w, agg);
        int Kc = SEQL * KCH * b.w;
        long long wt_n = (long long)Kc * b.w;
        k_wtrans<<<(unsigned)((wt_n + 255) / 256), 256, 0, stream>>>(
            par + b.convW, cwt, Kc, b.w);
        k_gemm_async<<<dim3((P + 63) / 64, b.w / 64), 128, 0, stream>>>(
            agg, cwt, t3, P, b.w, Kc);

        // out linear + residual
        k_bn_stats<<<b.w, 256, 0, stream>>>(t3, lv[lvl], P, b.w, bnsum, bnsq);
        k_bn_apply<<<(unsigned)((nw + 255) / 256), 256, 0, stream>>>(
            t3, t1, bnsum, bnsq, cnt, par + b.outg, par + b.outb, 0.1f, P, b.w);
        k_gemm<true><<<dim3((P + 63) / 64, b.out_c / 64), 128, 0, stream>>>(
            t1, par + b.outW, xnxt, identptr, P, b.out_c, b.w);
        { float* tmp = xcur; xcur = xnxt; xnxt = tmp; }

        if (i == 5) {
            k_gpool<<<dim3(1024, NGR), 256, 0, stream>>>(xcur, lb[lvl], lv[lvl], P, 1024, xg);
        } else if (i % 2 == 1) {
            int half = (lvl == 0) ? 150 : 75;
            int nseg = NGR * half;
            int Cc   = b.out_c;
            long long zn = (long long)nseg * Cc;
            k_zero<<<(unsigned)((zn + 255) / 256), 256, 0, stream>>>(xsum, zn);
            k_zero<<<(unsigned)((nseg * 3 + 255) / 256), 256, 0, stream>>>(psum, nseg * 3);
            k_zero<<<(unsigned)((nseg * 9 + 255) / 256), 256, 0, stream>>>(osum, nseg * 9);
            k_zero<<<(unsigned)((nseg + 255) / 256), 256, 0, stream>>>(csum, nseg);
            k_pool_acc<<<(P + 127) / 128, 128, 0, stream>>>(
                xcur, lp[lvl], ls[lvl], lo[lvl], lb[lvl], lv[lvl],
                P, Cc, half, xsum, psum, osum, csum);
            k_pool_fin<<<(nseg + 127) / 128, 128, 0, stream>>>(
                xnxt, lp[lvl + 1], ls[lvl + 1], lo[lvl + 1], lb[lvl + 1], lv[lvl + 1],
                xsum, psum, osum, csum, nseg, Cc, half);
            { float* tmp = xcur; xcur = xnxt; xnxt = tmp; }
        }
    }

    k_mlp<<<1, 256, 0, stream>>>(xg, par + oW1, par + oB1, par + oW2, par + oB2,
                                 par + oW3, par + oB3, (float*)d_out);
}
